// MultiHeadAttention_63118839382027
// MI455X (gfx1250) — matrix operand order repaired and checked by reading, not compiled, not run on hardware
//
#include <hip/hip_runtime.h>
#include <hip/hip_bf16.h>

typedef __bf16 bf16;
typedef __attribute__((ext_vector_type(16))) __bf16 v16bf;
typedef __attribute__((ext_vector_type(8)))  __bf16 v8bf;
typedef __attribute__((ext_vector_type(8)))  float  v8f;
typedef __attribute__((ext_vector_type(4)))  float  v4f;

#define BATCH 2
#define SEQ   2048
#define HID   2048
#define NH    16
#define DH    128
#define MTOT  (BATCH*SEQ)          // 4096 rows
#define SCALE 0.08838834764831845f // 1/sqrt(128)

__device__ __forceinline__ v8f wmma_bf16(v16bf a, v16bf b, v8f c) {
    return __builtin_amdgcn_wmma_f32_16x16x32_bf16(
        /*neg_a=*/false, a, /*neg_b=*/false, b,
        /*c_mod=*/(short)0, c, /*reuse_a=*/false, /*reuse_b=*/false);
}

__device__ __forceinline__ v16bf pack2(v8bf a0, v8bf a1) {
    v16bf r;
#pragma unroll
    for (int i = 0; i < 8; ++i) { r[i] = a0[i]; r[8 + i] = a1[i]; }
    return r;
}

// ---------------------------------------------------------------------------
// Kernel 0: one-pass f32 -> bf16 conversion (vectorized, 8 elems/thread)
// ---------------------------------------------------------------------------
__global__ __launch_bounds__(256) void cvt_bf16_kernel(
    const float* __restrict__ src, bf16* __restrict__ dst)
{
    const size_t i = ((size_t)blockIdx.x * blockDim.x + threadIdx.x) * 8;
    v4f x0 = *(const v4f*)(src + i);
    v4f x1 = *(const v4f*)(src + i + 4);
    v8bf o;
#pragma unroll
    for (int j = 0; j < 4; ++j) { o[j] = (__bf16)x0[j]; o[4 + j] = (__bf16)x1[j]; }
    *(v8bf*)(dst + i) = o;
}

// ---------------------------------------------------------------------------
// Kernel 1: P = X @ W^T, all-bf16 operands, f32 accumulate.
// One wave computes a 32(M) x 64(N) strip: 2 A-frags x 4 B-frags = 8 WMMAs
// per 32-wide k-step; B fragment shared by both A fragments.
// mode 0: out[((b*NH+h)*SEQ+s)*DH+d]  (Q,K head-major)
// mode 1: out[((b*NH+h)*DH+d)*SEQ+s]  (V transposed: d-major for PV B-frags)
// ---------------------------------------------------------------------------
__global__ __launch_bounds__(128) void proj_kernel(
    const bf16* __restrict__ X, const bf16* __restrict__ W,
    bf16* __restrict__ out, int mode)
{
    const int wave = (blockIdx.x * blockDim.x + threadIdx.x) >> 5;
    const int lane = threadIdx.x & 31;
    const int lm = lane & 15, lh = lane >> 4;

    const int NG = HID / 64;          // 32 n-groups
    const int mt = wave / NG;         // 0..127 (32-row tiles)
    const int ng = wave % NG;         // 0..31
    const int m0 = mt * 32;
    const int n0 = ng * 64;

    v8f acc[2][4] = {};
    const bf16* xrow0 = X + (size_t)(m0 + lm) * HID;
    const bf16* xrow1 = X + (size_t)(m0 + 16 + lm) * HID;

    for (int k0 = 0; k0 < HID; k0 += 32) {
        // A fragments: e<8 -> k=k0+8*lh+e ; e>=8 -> k=k0+16+8*lh+(e-8)
        v16bf a0 = pack2(*(const v8bf*)(xrow0 + k0 + lh * 8),
                         *(const v8bf*)(xrow0 + k0 + 16 + lh * 8));
        v16bf a1 = pack2(*(const v8bf*)(xrow1 + k0 + lh * 8),
                         *(const v8bf*)(xrow1 + k0 + 16 + lh * 8));
#pragma unroll
        for (int t = 0; t < 4; ++t) {
            // B fragment: B[k][n] = W[n][k]; lane col n0+16t+lm; e<8 -> k=k0+8*lh+e ; e>=8 -> k=k0+16+8*lh+(e-8)
            const bf16* wrow = W + (size_t)(n0 + t * 16 + lm) * HID + k0 + lh * 8;
            v16bf bfrag = pack2(*(const v8bf*)(wrow),
                                *(const v8bf*)(wrow + 16));
            acc[0][t] = wmma_bf16(a0, bfrag, acc[0][t]);
            acc[1][t] = wmma_bf16(a1, bfrag, acc[1][t]);
        }
    }

#pragma unroll
    for (int mi = 0; mi < 2; ++mi) {
#pragma unroll
        for (int t = 0; t < 4; ++t) {
            const int n = n0 + t * 16 + lm;
            const int h = n >> 7, d = n & 127;
#pragma unroll
            for (int r = 0; r < 8; ++r) {
                const int m = m0 + mi * 16 + r + lh * 8;
                const int b = m >> 11, s = m & (SEQ - 1);
                size_t idx = (mode == 0)
                    ? ((size_t)((b * NH + h) * SEQ + s)) * DH + d
                    : ((size_t)((b * NH + h) * DH + d)) * SEQ + s;
                out[idx] = (__bf16)acc[mi][t][r];
            }
        }
    }
}

// ---------------------------------------------------------------------------
// Kernel 2: causal flash attention per (b,h,16-query tile). Keys in steps
// of 32. Scores: 8 WMMAs. Online softmax with shfl_xor row reductions.
// P transposed C-layout -> A-fragment through LDS. PV: 8 WMMAs.
// ---------------------------------------------------------------------------
__global__ __launch_bounds__(128) void attn_kernel(
    const bf16* __restrict__ Q, const bf16* __restrict__ K,
    const bf16* __restrict__ Vt, bf16* __restrict__ HO)
{
    __shared__ bf16 pbuf[4 * 16 * 32];   // 1 KB per wave

    const int wave = (blockIdx.x * blockDim.x + threadIdx.x) >> 5;
    const int lane = threadIdx.x & 31;
    const int lm = lane & 15, lh = lane >> 4;

    const int bh = wave >> 7;            // 0..31 (b*16+h)
    const int qt = wave & 127;           // query tile
    const int q0 = qt * 16;

    const bf16* Qh  = Q  + (size_t)bh * SEQ * DH;
    const bf16* Kh  = K  + (size_t)bh * SEQ * DH;
    const bf16* Vth = Vt + (size_t)bh * DH * SEQ;
    const int b = bh >> 4, h = bh & 15;

    // Preload the 4 Q A-fragments (16 rows x 128 d)
    v16bf qf[4];
    const bf16* qrow = Qh + (size_t)(q0 + lm) * DH;
#pragma unroll
    for (int kk = 0; kk < 4; ++kk) {
        v8bf a0 = *(const v8bf*)(qrow + kk * 32 + lh * 8);
        v8bf a1 = *(const v8bf*)(qrow + kk * 32 + 16 + lh * 8);
        qf[kk] = pack2(a0, a1);
    }

    v8f o[8] = {};
    float rmax[8], rsum[8];
#pragma unroll
    for (int r = 0; r < 8; ++r) { rmax[r] = -INFINITY; rsum[r] = 0.0f; }

    bf16* pb = pbuf + (threadIdx.x >> 5) * (16 * 32);
    const int qmax = q0 + 15;

    for (int t0 = 0; t0 <= qmax; t0 += 32) {
        // ---- scores: 16 rows x 32 keys, accumulated over d=128 ----
        v8f sc[2] = {};
#pragma unroll
        for (int nt = 0; nt < 2; ++nt) {
            const bf16* krow = Kh + (size_t)(t0 + nt * 16 + lm) * DH + lh * 8;
#pragma unroll
            for (int kk = 0; kk < 4; ++kk) {
                v8bf b0 = *(const v8bf*)(krow + kk * 32);
                v8bf b1 = *(const v8bf*)(krow + kk * 32 + 16);
                sc[nt] = wmma_bf16(qf[kk], pack2(b0, b1), sc[nt]);
            }
        }

        // ---- scale + causal mask + online softmax ----
#pragma unroll
        for (int r = 0; r < 8; ++r) {
            const int q = q0 + r + lh * 8;
            float mrow = -INFINITY;
#pragma unroll
            for (int nt = 0; nt < 2; ++nt) {
                const int t = t0 + nt * 16 + lm;
                float s = sc[nt][r] * SCALE;
                if (t > q) s = -INFINITY;
                sc[nt][r] = s;
                mrow = fmaxf(mrow, s);
            }
#pragma unroll
            for (int off = 1; off < 16; off <<= 1)
                mrow = fmaxf(mrow, __shfl_xor(mrow, off, 32));
            const float nmax = fmaxf(rmax[r], mrow);
            const float scl = __expf(rmax[r] - nmax);
            rmax[r] = nmax;

            float psum = 0.0f;
#pragma unroll
            for (int nt = 0; nt < 2; ++nt) {
                float p = __expf(sc[nt][r] - nmax);
                sc[nt][r] = p;
                psum += p;
            }
#pragma unroll
            for (int off = 1; off < 16; off <<= 1)
                psum += __shfl_xor(psum, off, 32);
            rsum[r] = rsum[r] * scl + psum;
#pragma unroll
            for (int dt = 0; dt < 8; ++dt) o[dt][r] *= scl;
        }

        // ---- transpose P (C-layout -> A-fragment) through LDS ----
#pragma unroll
        for (int nt = 0; nt < 2; ++nt)
#pragma unroll
            for (int r = 0; r < 8; ++r)
                pb[(r + lh * 8) * 32 + nt * 16 + lm] = (__bf16)sc[nt][r];
        // same-wave DS ops are hardware-ordered; compiler inserts s_wait_dscnt
        v8bf p0 = *(const v8bf*)(pb + lm * 32 + lh * 8);
        v8bf p1 = *(const v8bf*)(pb + lm * 32 + 16 + lh * 8);
        v16bf pf = pack2(p0, p1);

        // ---- PV: (16x32) @ (32x128) into 8 resident accumulators ----
#pragma unroll
        for (int dt = 0; dt < 8; ++dt) {
            const bf16* vrow = Vth + (size_t)(dt * 16 + lm) * SEQ + t0 + lh * 8;
            v8bf b0 = *(const v8bf*)(vrow);
            v8bf b1 = *(const v8bf*)(vrow + 16);
            o[dt] = wmma_bf16(pf, pack2(b0, b1), o[dt]);
        }
    }

    // ---- finalize: divide by row sums, store HO[b][s][h*128+d] ----
#pragma unroll
    for (int r = 0; r < 8; ++r) {
        const int s = q0 + r + lh * 8;
        const float inv = 1.0f / rsum[r];
        bf16* orow = HO + ((size_t)(b * SEQ + s)) * HID + h * DH;
#pragma unroll
        for (int dt = 0; dt < 8; ++dt)
            orow[dt * 16 + lm] = (__bf16)(o[dt][r] * inv);
    }
}

// ---------------------------------------------------------------------------
// Kernel 3: out = HO @ Wo^T + bo  (HO bf16, Wo bf16, out f32). Same 32x64
// strip decomposition as proj_kernel.
// ---------------------------------------------------------------------------
__global__ __launch_bounds__(128) void outproj_kernel(
    const bf16* __restrict__ HOp, const bf16* __restrict__ W,
    const float* __restrict__ bo, float* __restrict__ out)
{
    const int wave = (blockIdx.x * blockDim.x + threadIdx.x) >> 5;
    const int lane = threadIdx.x & 31;
    const int lm = lane & 15, lh = lane >> 4;

    const int NG = HID / 64;
    const int mt = wave / NG;
    const int ng = wave % NG;
    const int m0 = mt * 32;
    const int n0 = ng * 64;

    v8f acc[2][4] = {};
    const bf16* hrow0 = HOp + (size_t)(m0 + lm) * HID;
    const bf16* hrow1 = HOp + (size_t)(m0 + 16 + lm) * HID;

    for (int k0 = 0; k0 < HID; k0 += 32) {
        v16bf a0 = pack2(*(const v8bf*)(hrow0 + k0 + lh * 8),
                         *(const v8bf*)(hrow0 + k0 + 16 + lh * 8));
        v16bf a1 = pack2(*(const v8bf*)(hrow1 + k0 + lh * 8),
                         *(const v8bf*)(hrow1 + k0 + 16 + lh * 8));
#pragma unroll
        for (int t = 0; t < 4; ++t) {
            const bf16* wrow = W + (size_t)(n0 + t * 16 + lm) * HID + k0 + lh * 8;
            v16bf bfrag = pack2(*(const v8bf*)(wrow),
                                *(const v8bf*)(wrow + 16));
            acc[0][t] = wmma_bf16(a0, bfrag, acc[0][t]);
            acc[1][t] = wmma_bf16(a1, bfrag, acc[1][t]);
        }
    }

#pragma unroll
    for (int mi = 0; mi < 2; ++mi) {
#pragma unroll
        for (int t = 0; t < 4; ++t) {
            const int n = n0 + t * 16 + lm;
            const float bias = bo[n];
#pragma unroll
            for (int r = 0; r < 8; ++r) {
                const int m = m0 + mi * 16 + r + lh * 8;
                out[(size_t)m * HID + n] = acc[mi][t][r] + bias;
            }
        }
    }
}

// ---------------------------------------------------------------------------
extern "C" void kernel_launch(void* const* d_in, const int* in_sizes, int n_in,
                              void* d_out, int out_size, void* d_ws, size_t ws_size,
                              hipStream_t stream) {
    const float* X  = (const float*)d_in[0];
    const float* Wq = (const float*)d_in[1];
    const float* Wk = (const float*)d_in[2];
    const float* Wv = (const float*)d_in[3];
    const float* Wo = (const float*)d_in[4];
    const float* bo = (const float*)d_in[5];
    float* out = (float*)d_out;

    const size_t XE = (size_t)MTOT * HID;   // 8,388,608
    const size_t WE = (size_t)HID * HID;    // 4,194,304

    bf16* Xbf  = (bf16*)d_ws;
    bf16* Wqbf = Xbf  + XE;
    bf16* Wkbf = Wqbf + WE;
    bf16* Wvbf = Wkbf + WE;
    bf16* Wobf = Wvbf + WE;
    bf16* Qw   = Wobf + WE;
    bf16* Kw   = Qw + XE;
    bf16* Vw   = Kw + XE;
    bf16* HOw  = Vw + XE;                   // total ~117 MiB of ws

    // ---- one-pass f32 -> bf16 conversions (removes cvt from GEMM loops) ----
    cvt_bf16_kernel<<<XE / (256 * 8), 256, 0, stream>>>(X,  Xbf);
    cvt_bf16_kernel<<<WE / (256 * 8), 256, 0, stream>>>(Wq, Wqbf);
    cvt_bf16_kernel<<<WE / (256 * 8), 256, 0, stream>>>(Wk, Wkbf);
    cvt_bf16_kernel<<<WE / (256 * 8), 256, 0, stream>>>(Wv, Wvbf);
    cvt_bf16_kernel<<<WE / (256 * 8), 256, 0, stream>>>(Wo, Wobf);

    dim3 blk(128);
    // 128 m-tiles * 32 n-groups = 4096 waves / 4 per block
    proj_kernel<<<1024, blk, 0, stream>>>(Xbf, Wqbf, Qw, 0);
    proj_kernel<<<1024, blk, 0, stream>>>(Xbf, Wkbf, Kw, 0);
    proj_kernel<<<1024, blk, 0, stream>>>(Xbf, Wvbf, Vw, 1);
    // 32 (b,h) * 128 q-tiles = 4096 waves
    attn_kernel<<<1024, blk, 0, stream>>>(Qw, Kw, Vw, HOw);
    outproj_kernel<<<1024, blk, 0, stream>>>(HOw, Wobf, bo, out);
}
